// Local_align_23278722744898
// MI455X (gfx1250) — compile-verified, matrix operand order checked by reading, not run on hardware
//
#include <hip/hip_runtime.h>
#include <hip/hip_bf16.h>

// ---------------- problem constants ----------------
#define Bn   8
#define NCn  150
#define NCP  160     // NC padded to 16
#define WD   768
#define Cc   256
#define HH   128
#define WW   128
#define OUTD 2305    // C*9 + 1
#define KWt  2304    // C*9
#define CHS  40      // LDS channel stride (80B: 16B-aligned, bank-stride 20 -> conflict free)

typedef __attribute__((ext_vector_type(16))) __bf16 v16bf;
typedef __attribute__((ext_vector_type(8)))  float  v8f;

union Frag { uint4 u[2]; v16bf v; };

__device__ __forceinline__ v8f wmma_bf16(const Frag& a, const Frag& b, v8f c) {
    return __builtin_amdgcn_wmma_f32_16x16x32_bf16(false, a.v, false, b.v,
                                                   (short)0, c, false, false);
}

__device__ __forceinline__ v8f vzero8() {
    v8f z;
#pragma unroll
    for (int i = 0; i < 8; ++i) z[i] = 0.0f;
    return z;
}

__device__ __forceinline__ void split_bf16(float x, __bf16& hi, __bf16& lo) {
    hi = (__bf16)x;
    lo = (__bf16)(x - (float)hi);
}

// ---- CDNA5 async global->LDS copy (ASYNCcnt path), 16B per lane ----
typedef __attribute__((address_space(3))) const void lds_cvoid_t;

__device__ __forceinline__ unsigned lds_byte_addr(const void* p) {
    return (unsigned)(size_t)(lds_cvoid_t*)p;   // generic -> LDS offset
}

__device__ __forceinline__ void async_ld_b128(unsigned lds_byte, const void* g) {
    asm volatile("global_load_async_to_lds_b128 %0, %1, off"
                 :: "v"(lds_byte), "v"((unsigned long long)(size_t)g)
                 : "memory");
}

__device__ __forceinline__ void wait_async0() {
    asm volatile("s_wait_asynccnt 0x0" ::: "memory");
}

// ---------------- kernel 0: Wt f32 -> bf16 hi/lo planes ----------------
__global__ void k_wt_split(const float* __restrict__ Wt,
                           __bf16* __restrict__ wtH, __bf16* __restrict__ wtL, int n) {
    int i = blockIdx.x * 256 + threadIdx.x;
    if (i < n) {
        __bf16 hi, lo; split_bf16(Wt[i], hi, lo);
        wtH[i] = hi; wtL[i] = lo;
    }
}

// ---------------- kernel 1: tf = l2norm(text) @ Wt^T  (bias bt added later) ----
// grid: (10 nc-tiles, 8 batches), 256 threads (8 waves). WMMA bf16 hi/lo (3 products).
__global__ void __launch_bounds__(256)
k_text_gemm(const float* __restrict__ text,
            const __bf16* __restrict__ wtH, const __bf16* __restrict__ wtL,
            float* __restrict__ tf) {
    __shared__ __bf16 sAH[16 * 776];   // row stride 776 elems -> bank-conflict-free frag loads
    __shared__ __bf16 sAL[16 * 776];

    const int nct = blockIdx.x, b = blockIdx.y;
    const int nc0 = nct * 16;
    const int tid = threadIdx.x;
    const int wv = tid >> 5, lane = tid & 31;
    const int lane16 = lane & 15;
    const int khalf = (lane >> 4) << 3;           // 0 or 8

    // ---- stage normalized text rows (2 rows per wave) as bf16 hi/lo ----
#pragma unroll
    for (int rr = 0; rr < 2; ++rr) {
        const int m = wv * 2 + rr;
        const int nc = nc0 + m;
        float vals[24];
        float s = 0.0f;
        if (nc < NCn) {
            const float* tp = text + ((size_t)(b * NCn + nc)) * WD;
#pragma unroll
            for (int i = 0; i < 24; ++i) { float v = tp[lane + i * 32]; vals[i] = v; s += v * v; }
        } else {
#pragma unroll
            for (int i = 0; i < 24; ++i) vals[i] = 0.0f;
        }
#pragma unroll
        for (int off = 16; off >= 1; off >>= 1) s += __shfl_xor(s, off, 32);
        const float inv = 1.0f / fmaxf(sqrtf(s), 1e-12f);
#pragma unroll
        for (int i = 0; i < 24; ++i) {
            __bf16 hi, lo; split_bf16(vals[i] * inv, hi, lo);
            sAH[m * 776 + lane + i * 32] = hi;
            sAL[m * 776 + lane + i * 32] = lo;
        }
    }
    __syncthreads();

    // ---- each wave loops over 16-wide output-column tiles of OUTD ----
    for (int tile = wv; tile < 145; tile += 8) {
        const int j0 = tile * 16;
        const int jc = min(j0 + lane16, OUTD - 1);
        const __bf16* bRowH = wtH + (size_t)jc * WD;
        const __bf16* bRowL = wtL + (size_t)jc * WD;

        v8f acc = vzero8();
#pragma unroll 4
        for (int ks = 0; ks < 24; ++ks) {
            const int k0 = ks * 32;
            Frag aH, aL, bH, bL;
            const __bf16* arH = sAH + lane16 * 776 + k0 + khalf;
            const __bf16* arL = sAL + lane16 * 776 + k0 + khalf;
            aH.u[0] = *(const uint4*)(arH);      aH.u[1] = *(const uint4*)(arH + 16);
            aL.u[0] = *(const uint4*)(arL);      aL.u[1] = *(const uint4*)(arL + 16);
            const __bf16* brH = bRowH + k0 + khalf;
            const __bf16* brL = bRowL + k0 + khalf;
            bH.u[0] = *(const uint4*)(brH);      bH.u[1] = *(const uint4*)(brH + 16);
            bL.u[0] = *(const uint4*)(brL);      bL.u[1] = *(const uint4*)(brL + 16);
            acc = wmma_bf16(aH, bH, acc);
            acc = wmma_bf16(aH, bL, acc);
            acc = wmma_bf16(aL, bH, acc);
        }
        const int colj = j0 + lane16;
        if (colj < OUTD) {
            const int mbase = (lane < 16) ? 0 : 8;
#pragma unroll
            for (int r = 0; r < 8; ++r)
                tf[(size_t)(b * NCP + nc0 + mbase + r) * OUTD + colj] = acc[r];
        }
    }
}

// ---------------- kernel 2: softmax over taps + relayout to A-weights ------
// grid: B*NCP blocks, 256 threads (thread = channel). k' = tap*256 + c.
__global__ void k_softmax_relayout(const float* __restrict__ tf, const float* __restrict__ bt,
                                   __bf16* __restrict__ wAH, __bf16* __restrict__ wAL,
                                   float* __restrict__ biasW) {
    const int bn = blockIdx.x;          // b*NCP + nc
    const int nc = bn % NCP;
    const int c = threadIdx.x;
    const size_t wbase = (size_t)bn * KWt;
    if (nc >= NCn) {
#pragma unroll
        for (int t = 0; t < 9; ++t) {
            wAH[wbase + t * Cc + c] = (__bf16)0.0f;
            wAL[wbase + t * Cc + c] = (__bf16)0.0f;
        }
        if (c == 0) biasW[bn] = 0.0f;
        return;
    }
    const size_t base = (size_t)bn * OUTD;
    float v[9]; float m = -1e30f;
#pragma unroll
    for (int t = 0; t < 9; ++t) {
        v[t] = tf[base + c * 9 + t] + bt[c * 9 + t];
        m = fmaxf(m, v[t]);
    }
    float s = 0.0f;
#pragma unroll
    for (int t = 0; t < 9; ++t) { v[t] = expf(v[t] - m); s += v[t]; }
    const float invs = 1.0f / s;
#pragma unroll
    for (int t = 0; t < 9; ++t) {
        __bf16 hi, lo; split_bf16(v[t] * invs, hi, lo);
        wAH[wbase + t * Cc + c] = hi;
        wAL[wbase + t * Cc + c] = lo;
    }
    if (c == 0) biasW[bn] = tf[base + KWt] + bt[KWt];
}

// ---------------- kernel 3: vision l2norm over C + bf16 hi/lo, channel-innermost ---
// grid: (4 w-tiles, 128 h, 8 b), 256 threads. out layout: [b][h][x][c]
__global__ void k_vision_norm(const float* __restrict__ vision,
                              __bf16* __restrict__ imgH, __bf16* __restrict__ imgL) {
    __shared__ float tile[256 * 33];
    __shared__ float sinv[32];
    const int wt = blockIdx.x, h = blockIdx.y, b = blockIdx.z;
    const int tid = threadIdx.x;
    const int x = tid & 31;
    const int cb = tid >> 5;
#pragma unroll 4
    for (int i = 0; i < 32; ++i) {
        const int c = i * 8 + cb;
        tile[c * 33 + x] = vision[(((size_t)b * Cc + c) * HH + h) * WW + wt * 32 + x];
    }
    __syncthreads();
    if (tid < 32) {
        float s = 0.0f;
        for (int c = 0; c < Cc; ++c) { float v = tile[c * 33 + tid]; s += v * v; }
        sinv[tid] = 1.0f / fmaxf(sqrtf(s), 1e-12f);
    }
    __syncthreads();
#pragma unroll 4
    for (int i = 0; i < 32; ++i) {
        const int idx = i * 256 + tid;
        const int xx = idx >> 8;
        const int c = idx & 255;
        __bf16 hi, lo; split_bf16(tile[c * 33 + xx] * sinv[xx], hi, lo);
        const size_t dst = (((size_t)b * HH + h) * WW + wt * 32 + xx) * Cc + c;
        imgH[dst] = hi; imgL[dst] = lo;
    }
}

// ---------------- kernel 4: implicit-GEMM conv, WMMA bf16 hi/lo -----------
// grid: (128 h, 8 b), 256 threads = 8 waves. Each wave owns 16 pixel columns.
// Per block: all 10 NC-tiles accumulate in VGPRs; K loop = 8 ch-blocks x 9 taps.
// Image rows are staged into LDS with GLOBAL_LOAD_ASYNC_TO_LDS_B128 (ASYNCcnt).
__global__ void __launch_bounds__(256)
k_conv(const __bf16* __restrict__ imgH, const __bf16* __restrict__ imgL,
       const __bf16* __restrict__ wAH, const __bf16* __restrict__ wAL,
       const float* __restrict__ biasW, float* __restrict__ out) {
    __shared__ __bf16 ldsH[3 * 130 * CHS];   // [row 3][px 130][ch 40(32 used)]
    __shared__ __bf16 ldsL[3 * 130 * CHS];
    __shared__ float  ldsBias[NCP];

    const int h = blockIdx.x, b = blockIdx.y;
    const int tid = threadIdx.x;
    const int wv = tid >> 5, lane = tid & 31;
    const int lane16 = lane & 15;
    const int khalf = (lane >> 4) << 3;        // 0 or 8

    if (tid < NCP) ldsBias[tid] = biasW[b * NCP + tid];

    const unsigned ldsHbase = lds_byte_addr(ldsH);
    const unsigned ldsLbase = lds_byte_addr(ldsL);

    v8f acc[10];
#pragma unroll
    for (int i = 0; i < 10; ++i) acc[i] = vzero8();

    for (int cb = 0; cb < 8; ++cb) {
        __syncthreads();   // prior-iteration LDS reads complete before overwrite
        // ---- stage 3 rows x 130 px x 32 ch (hi+lo) into LDS, 16B async chunks ----
        const int c0 = cb * 32;
        for (int q = tid; q < 3 * 130 * 4; q += 256) {
            const int r = q / (130 * 4);
            const int rem = q - r * (130 * 4);
            const int px = rem >> 2;
            const int part = rem & 3;
            const int xr = px - 1;
            const int hr = h + r - 1;
            const unsigned dstB = (unsigned)(((r * 130 + px) * CHS + part * 8) * 2);
            if (xr >= 0 && xr < WW && hr >= 0 && hr < HH) {
                const size_t src = (((size_t)b * HH + hr) * WW + xr) * Cc + c0 + part * 8;
                async_ld_b128(ldsHbase + dstB, imgH + src);
                async_ld_b128(ldsLbase + dstB, imgL + src);
            } else {
                const uint4 z = make_uint4(0, 0, 0, 0);
                *(uint4*)((char*)ldsH + dstB) = z;
                *(uint4*)((char*)ldsL + dstB) = z;
            }
        }
        wait_async0();     // my async copies have landed in LDS
        __syncthreads();   // ... and everyone else's too

        const size_t abase0 = ((size_t)b * NCP) * KWt + c0;
        for (int tap = 0; tap < 9; ++tap) {
            const int dh = tap / 3, dw = tap % 3;
            const int px = wv * 16 + lane16 + dw;            // x + dw - 1, +1 halo shift
            const __bf16* brH = ldsH + (dh * 130 + px) * CHS + khalf;
            const __bf16* brL = ldsL + (dh * 130 + px) * CHS + khalf;
            Frag bH, bL;
            bH.u[0] = *(const uint4*)(brH);  bH.u[1] = *(const uint4*)(brH + 16);
            bL.u[0] = *(const uint4*)(brL);  bL.u[1] = *(const uint4*)(brL + 16);

            const size_t abase = abase0 + (size_t)tap * Cc;
#pragma unroll
            for (int nct = 0; nct < 10; ++nct) {
                const size_t arow = abase + (size_t)(nct * 16 + lane16) * KWt + khalf;
                Frag aH, aL;
                aH.u[0] = *(const uint4*)(wAH + arow);  aH.u[1] = *(const uint4*)(wAH + arow + 16);
                aL.u[0] = *(const uint4*)(wAL + arow);  aL.u[1] = *(const uint4*)(wAL + arow + 16);
                acc[nct] = wmma_bf16(aH, bH, acc[nct]);
                acc[nct] = wmma_bf16(aH, bL, acc[nct]);
                acc[nct] = wmma_bf16(aL, bH, acc[nct]);
            }
        }
    }

    // ---- epilogue: + bias, store out[b][nc][h][x] ----
    const int x = wv * 16 + lane16;
    const int mbase = (lane < 16) ? 0 : 8;
#pragma unroll
    for (int nct = 0; nct < 10; ++nct) {
#pragma unroll
        for (int r = 0; r < 8; ++r) {
            const int nc = nct * 16 + mbase + r;
            if (nc < NCn)
                out[(((size_t)b * NCn + nc) * HH + h) * WW + x] = acc[nct][r] + ldsBias[nc];
        }
    }
}

// ---------------- host launcher ----------------
extern "C" void kernel_launch(void* const* d_in, const int* in_sizes, int n_in,
                              void* d_out, int out_size, void* d_ws, size_t ws_size,
                              hipStream_t stream) {
    (void)in_sizes; (void)n_in; (void)out_size; (void)ws_size;
    const float* vision = (const float*)d_in[0];
    const float* text   = (const float*)d_in[1];
    const float* Wt     = (const float*)d_in[2];
    const float* bt     = (const float*)d_in[3];
    float* out = (float*)d_out;

    char* ws = (char*)d_ws;
    size_t off = 0;
    auto alloc = [&](size_t bytes) -> char* {
        char* p = ws + off;
        off = (off + bytes + 255) & ~(size_t)255;
        return p;
    };
    const size_t nWt = (size_t)OUTD * WD;
    __bf16* wtH  = (__bf16*)alloc(nWt * 2);
    __bf16* wtL  = (__bf16*)alloc(nWt * 2);
    float*  tf   = (float*) alloc((size_t)Bn * NCP * OUTD * 4);
    __bf16* wAH  = (__bf16*)alloc((size_t)Bn * NCP * KWt * 2);
    __bf16* wAL  = (__bf16*)alloc((size_t)Bn * NCP * KWt * 2);
    float*  biasW= (float*) alloc((size_t)Bn * NCP * 4);
    __bf16* imgH = (__bf16*)alloc((size_t)Bn * HH * WW * Cc * 2);
    __bf16* imgL = (__bf16*)alloc((size_t)Bn * HH * WW * Cc * 2);

    k_wt_split<<<(int)((nWt + 255) / 256), 256, 0, stream>>>(Wt, wtH, wtL, (int)nWt);
    k_text_gemm<<<dim3(10, Bn), 256, 0, stream>>>(text, wtH, wtL, tf);
    k_softmax_relayout<<<Bn * NCP, 256, 0, stream>>>(tf, bt, wAH, wAL, biasW);
    k_vision_norm<<<dim3(4, HH, Bn), 256, 0, stream>>>(vision, imgH, imgL);
    k_conv<<<dim3(HH, Bn), 256, 0, stream>>>(imgH, imgL, wAH, wAL, biasW, out);
}
